// GaussianKernel_18631568130545
// MI455X (gfx1250) — matrix operand order repaired and checked by reading, not compiled, not run on hardware
//
#include <hip/hip_runtime.h>
#include <hip/hip_bf16.h>
#include <math.h>

// Gaussian RBF kernel matrix: out[b,n,m] = exp(c * (|Q_n|^2 + |KV_m|^2 - 2 Q_n.KV_m))
// c = -0.5 * exp(-2*log_sigmas[0]).
// Q: [B,N,D] f32, KV: [B,D,M] f32, out: [B,N,M] f32.  B=16, N=M=2048, D=128.

typedef __attribute__((ext_vector_type(16))) __bf16 bf16x16;
typedef __attribute__((ext_vector_type(8)))  __bf16 bf16x8;
typedef __attribute__((ext_vector_type(4)))  __bf16 bf16x4;
typedef __attribute__((ext_vector_type(8)))  float  f32x8;
typedef __attribute__((ext_vector_type(4)))  float  f32x4;

#define N_DIM 2048
#define M_DIM 2048
#define D_DIM 128
#define TILE  128
#define KC    32
#define LSTR  40   // 32 + 8 bf16 pad -> 80B row stride: conflict-free b128 fragment loads

__global__ __launch_bounds__(256)
void gaussian_wmma_bf16x3(const float* __restrict__ Q,
                          const float* __restrict__ KV,
                          const float* __restrict__ log_sigmas,
                          float* __restrict__ out)
{
    __shared__ __bf16 sQhi[TILE * LSTR];
    __shared__ __bf16 sQlo[TILE * LSTR];
    __shared__ __bf16 sKhi[TILE * LSTR];
    __shared__ __bf16 sKlo[TILE * LSTR];
    __shared__ float  sQsq[TILE];
    __shared__ float  sKsq[TILE];

    const int tid  = threadIdx.x;
    const int lane = tid & 31;
    const int wave = tid >> 5;
    const int hiL  = lane >> 4;      // lane group 0/1
    const int lo16 = lane & 15;

    const int b  = blockIdx.z;
    const int n0 = blockIdx.y * TILE;
    const int m0 = blockIdx.x * TILE;

    // ---------------- exact fp32 norms straight from global ----------------
    if (tid < 128) {
        const float* q = Q + ((size_t)(b * N_DIM + n0 + tid)) * D_DIM;
        float s = 0.f;
        #pragma unroll
        for (int i = 0; i < D_DIM / 4; ++i) {
            f32x4 v = ((const f32x4*)q)[i];
            s += v.x * v.x + v.y * v.y + v.z * v.z + v.w * v.w;
        }
        sQsq[tid] = s;
    } else {
        const int m = tid - 128;
        const float* k = KV + ((size_t)b * D_DIM) * M_DIM + (m0 + m);
        float s = 0.f;
        #pragma unroll 8
        for (int d = 0; d < D_DIM; ++d) {
            float v = k[(size_t)d * M_DIM];
            s += v * v;
        }
        sKsq[m] = s;
    }

    // ---------------- accumulators: wave owns 32(n) x 64(m) ----------------
    const int wr = wave >> 1;   // 0..3 -> n sub-block of 32 rows
    const int wc = wave & 1;    // 0..1 -> m sub-block of 64 cols

    f32x8 acc[2][4];
    #pragma unroll
    for (int i = 0; i < 2; ++i)
        #pragma unroll
        for (int j = 0; j < 4; ++j)
            #pragma unroll
            for (int v = 0; v < 8; ++v)
                acc[i][j][v] = 0.0f;

    // ---------------- K loop: stage 32-deep chunk, split f32 -> bf16 hi/lo ----------------
    for (int ck = 0; ck < D_DIM; ck += KC) {
        // Q slice [128 x 32], row-major K-contiguous in LDS
        {
            const int row = tid >> 1;
            const int cb  = (tid & 1) * 16;
            const float* gq = Q + ((size_t)(b * N_DIM + n0 + row)) * D_DIM + ck + cb;
            #pragma unroll
            for (int i = 0; i < 4; ++i) {
                f32x4 v = ((const f32x4*)gq)[i];
                bf16x4 h, l;
                #pragma unroll
                for (int e = 0; e < 4; ++e) {
                    float x  = v[e];
                    __bf16 hb = (__bf16)x;              // RTNE high part
                    h[e] = hb;
                    l[e] = (__bf16)(x - (float)hb);     // residual low part
                }
                const int off = row * LSTR + cb + 4 * i;
                *(bf16x4*)&sQhi[off] = h;
                *(bf16x4*)&sQlo[off] = l;
            }
            // gfx1250 prefetch of next chunk while WMMAs run
            if (ck + KC < D_DIM) __builtin_prefetch(gq + KC, 0, 0);
        }
        // KV slice [32 x 128] transposed into sK[m][k] (K-contiguous for B fragments)
        {
            const int m  = tid & 127;
            const int dh = (tid >> 7) * 16;
            const float* gk = KV + ((size_t)(b * D_DIM + ck)) * M_DIM + (m0 + m);
            #pragma unroll
            for (int i = 0; i < 4; ++i) {
                const int dl = dh + 4 * i;
                bf16x4 h, l;
                #pragma unroll
                for (int e = 0; e < 4; ++e) {
                    float x  = gk[(size_t)(dl + e) * M_DIM];
                    __bf16 hb = (__bf16)x;
                    h[e] = hb;
                    l[e] = (__bf16)(x - (float)hb);
                }
                const int off = m * LSTR + dl;
                *(bf16x4*)&sKhi[off] = h;
                *(bf16x4*)&sKlo[off] = l;
            }
            if (ck + KC < D_DIM) __builtin_prefetch(gk + (size_t)KC * M_DIM, 0, 0);
        }
        __syncthreads();

        // A fragments: 16x32 bf16 per ISA layout
        //   lanes 0-15: V0-3 = K 0..7, V4-7 = K 16..23 ; lanes 16-31: +8 / +24
        bf16x16 Ahi[2], Alo[2];
        #pragma unroll
        for (int i = 0; i < 2; ++i) {
            const int r    = (wr * 2 + i) * 16 + lo16;
            const int base = r * LSTR + 8 * hiL;
            bf16x8 a0 = *(const bf16x8*)&sQhi[base];
            bf16x8 a1 = *(const bf16x8*)&sQhi[base + 16];
            bf16x8 b0 = *(const bf16x8*)&sQlo[base];
            bf16x8 b1 = *(const bf16x8*)&sQlo[base + 16];
            #pragma unroll
            for (int e = 0; e < 8; ++e) {
                Ahi[i][e] = a0[e]; Ahi[i][e + 8] = a1[e];
                Alo[i][e] = b0[e]; Alo[i][e + 8] = b1[e];
            }
        }

        // B fragments + bf16x3 WMMA: Ahi*Bhi + Ahi*Blo + Alo*Bhi
        #pragma unroll
        for (int j = 0; j < 4; ++j) {
            const int r    = (wc * 4 + j) * 16 + lo16;  // lane <-> N column
            const int base = r * LSTR + 8 * hiL;        // lanes 0-15: K 0..7,16..23; 16-31: K 8..15,24..31
            bf16x16 Bhi, Blo;
            {
                bf16x8 c0 = *(const bf16x8*)&sKhi[base];
                bf16x8 c1 = *(const bf16x8*)&sKhi[base + 16];
                bf16x8 d0 = *(const bf16x8*)&sKlo[base];
                bf16x8 d1 = *(const bf16x8*)&sKlo[base + 16];
                #pragma unroll
                for (int e = 0; e < 8; ++e) {
                    Bhi[e] = c0[e]; Bhi[e + 8] = c1[e];
                    Blo[e] = d0[e]; Blo[e + 8] = d1[e];
                }
            }
            #pragma unroll
            for (int i = 0; i < 2; ++i) {
                acc[i][j] = __builtin_amdgcn_wmma_f32_16x16x32_bf16(
                    false, Ahi[i], false, Bhi, (short)0, acc[i][j], false, false);
                acc[i][j] = __builtin_amdgcn_wmma_f32_16x16x32_bf16(
                    false, Ahi[i], false, Blo, (short)0, acc[i][j], false, false);
                acc[i][j] = __builtin_amdgcn_wmma_f32_16x16x32_bf16(
                    false, Alo[i], false, Bhi, (short)0, acc[i][j], false, false);
            }
        }
        __syncthreads();
    }

    // ---------------- fused Gaussian epilogue ----------------
    const float cc = -0.5f * __expf(-2.0f * log_sigmas[0]);
    const size_t outb = (size_t)b * N_DIM * M_DIM;

    #pragma unroll
    for (int i = 0; i < 2; ++i) {
        #pragma unroll
        for (int j = 0; j < 4; ++j) {
            const int mL = wc * 64 + j * 16 + lo16;
            const float kn = sKsq[mL];
            #pragma unroll
            for (int v = 0; v < 8; ++v) {
                // D layout: VGPR v -> row v (lanes 0-15) or v+8 (lanes 16-31)
                const int nL = wr * 32 + i * 16 + v + 8 * hiL;
                const float d2 = sQsq[nL] + kn - 2.0f * acc[i][j][v];
                out[outb + (size_t)(n0 + nL) * M_DIM + (m0 + mL)] = __expf(cc * d2);
            }
        }
    }
}

extern "C" void kernel_launch(void* const* d_in, const int* in_sizes, int n_in,
                              void* d_out, int out_size, void* d_ws, size_t ws_size,
                              hipStream_t stream) {
    const float* Q  = (const float*)d_in[0];
    const float* KV = (const float*)d_in[1];
    const float* ls = (const float*)d_in[2];
    float* out = (float*)d_out;

    const int B = in_sizes[0] / (N_DIM * D_DIM);   // 16 for the reference shapes
    dim3 grid(M_DIM / TILE, N_DIM / TILE, B);
    gaussian_wmma_bf16x3<<<grid, 256, 0, stream>>>(Q, KV, ls, out);
}
